// BootTeacher_721554506540
// MI455X (gfx1250) — matrix operand order repaired and checked by reading, not compiled, not run on hardware
//
#include <hip/hip_runtime.h>
#include <math.h>

#define E_N     200000
#define P_N     50000
#define D_N     128
#define NNZ_N   2000000
#define C_N     32
#define NSEED_N 320

typedef __bf16 v16bf __attribute__((ext_vector_type(16)));
typedef float  v8f   __attribute__((ext_vector_type(8)));
typedef int    v8i   __attribute__((ext_vector_type(8)));

__device__ __forceinline__ v16bf as_v16bf(v8i x) {
    union { v8i i; v16bf b; } u; u.i = x; return u.b;
}

// round-to-nearest-even f32 -> bf16 (packed pair)
__device__ __forceinline__ unsigned short f2bf(float f) {
    unsigned int u = __float_as_uint(f);
    unsigned int r = u + 0x7FFFu + ((u >> 16) & 1u);
    return (unsigned short)(r >> 16);
}
__device__ __forceinline__ unsigned int pk_bf(float a, float b) {
    return (unsigned int)f2bf(a) | ((unsigned int)f2bf(b) << 16);
}

// ---------------------------------------------------------------- utility
__global__ void zero_f4(float4* __restrict__ p, long n4) {
    long i = (long)blockIdx.x * blockDim.x + threadIdx.x;
    long s = (long)gridDim.x * blockDim.x;
    float4 z = make_float4(0.f, 0.f, 0.f, 0.f);
    for (; i < n4; i += s) p[i] = z;
}

__global__ void copy_f4(const float4* __restrict__ src, float4* __restrict__ dst, long n4) {
    long i = (long)blockIdx.x * blockDim.x + threadIdx.x;
    long s = (long)gridDim.x * blockDim.x;
    for (; i < n4; i += s) dst[i] = src[i];
}

// ------------------------------------------------- weight -> B-fragment bf16
// W is (384,128) row-major f32 (gx = x @ W^T).  B tile = W^T[k0:k0+32, j0:j0+16].
// Per-lane B layout as packed here (contiguous k, same order as A): lane l: n = l%16, kbase = (l<16?0:16);
// vgpr v holds K = kbase + 2v, 2v+1.
// Fragment-contiguous packing: Wf[((jt*4+kt)*32 + lane)*8 + v]  -> each lane
// loads its whole fragment as 32 consecutive bytes (2x global_load_b128).
__global__ void conv_wfrag(const float* __restrict__ W, unsigned int* __restrict__ Wf) {
    int tid = blockIdx.x * blockDim.x + threadIdx.x;   // 24576 total
    if (tid >= (384 * 128) / 2) return;
    int v  = tid & 7;
    int l  = (tid >> 3) & 31;
    int kt = (tid >> 8) & 3;
    int jt = tid >> 10;
    int j = jt * 16 + (l & 15);
    int k = kt * 32 + ((l < 16) ? 0 : 16) + 2 * v;
    Wf[tid] = pk_bf(W[j * 128 + k], W[j * 128 + k + 1]);
}

// ---------------------------------------------------------------- seed mean
__global__ void seed_mean_k(const int* __restrict__ sidx, const float* __restrict__ es,
                            float* __restrict__ mean) {
    int d = threadIdx.x;  // 128 threads, 1 block
    float s = 0.f;
    for (int i = 0; i < NSEED_N; ++i) {
        int r = sidx[i];
        s += es[(long)r * D_N + d];
    }
    mean[d] = s * (1.0f / NSEED_N);
}

// ---------------------------------------------------------------- scatter-add
// wave-per-edge: lane handles 4 consecutive floats of the 128-wide row
__global__ void __launch_bounds__(256)
scatter_add_k(const float* __restrict__ src, const int* __restrict__ si,
              const int* __restrict__ di, const float* __restrict__ w,
              float* __restrict__ dst, int nnz) {
    int wave = threadIdx.x >> 5;
    int lane = threadIdx.x & 31;
    int e = blockIdx.x * 8 + wave;
    if (e >= nnz) return;
    int   r  = si[e];
    int   c  = di[e];
    float wv = w[e];
    const float4* sp = (const float4*)(src + (long)r * D_N);
    float* dp = dst + (long)c * D_N + lane * 4;
    __builtin_prefetch(dp, 1, 0);          // global_prefetch_b8 of the RMW line
    float4 v = sp[lane];
    atomicAdd(dp + 0, v.x * wv);
    atomicAdd(dp + 1, v.y * wv);
    atomicAdd(dp + 2, v.z * wv);
    atomicAdd(dp + 3, v.w * wv);
}

// ---------------------------------------------------------------- B fragment
// One lane's fragment = 8 consecutive dwords -> 2x global_load_b128.
__device__ __forceinline__ v16bf load_bfrag(const unsigned int* __restrict__ Wf,
                                            int jt, int kt, int lane) {
    const v8i* p = (const v8i*)(Wf + (((jt * 4 + kt) * 32) + lane) * 8);
    return as_v16bf(*p);
}

// ---------------------------------------------------------------- fused GRU
// Block = 256 threads = 8 waves, owns a 16-row tile of x/h.
// Wave w owns cols [w*16, w*16+16) of D, and computes 6 accumulators:
// (gx,gh) x (r,z,n).  Gates evaluated in registers in the WMMA C/D layout.
__global__ void __launch_bounds__(256)
gru_wmma_k(const float* __restrict__ x, float* __restrict__ h,
           const unsigned int* __restrict__ WihF, const unsigned int* __restrict__ WhhF,
           const float* __restrict__ bih, const float* __restrict__ bhh,
           const float* __restrict__ mean, int N) {
    __shared__ unsigned short Ax[16 * 128];
    __shared__ unsigned short Ah[16 * 128];

    const int t    = threadIdx.x;
    const int row0 = blockIdx.x * 16;

    // ---- stage A tiles (bf16), folding +mean into x for the e-GRU ----
    {
        int r  = t >> 4;            // 0..15
        int db = (t & 15) * 8;      // 0..120 step 8
        float4 x0, x1, h0, h1;
        if (row0 + r < N) {
            long g = (long)(row0 + r) * D_N + db;
            x0 = *(const float4*)(x + g);
            x1 = *(const float4*)(x + g + 4);
            h0 = *(const float4*)(h + g);
            h1 = *(const float4*)(h + g + 4);
            if (mean) {
                float4 m0 = *(const float4*)(mean + db);
                float4 m1 = *(const float4*)(mean + db + 4);
                x0.x += m0.x; x0.y += m0.y; x0.z += m0.z; x0.w += m0.w;
                x1.x += m1.x; x1.y += m1.y; x1.z += m1.z; x1.w += m1.w;
            }
        } else {
            x0 = x1 = h0 = h1 = make_float4(0.f, 0.f, 0.f, 0.f);
        }
        // pack 8 bf16 -> uint4, one ds_store_b128 per matrix per thread
        uint4 px, ph;
        px.x = pk_bf(x0.x, x0.y); px.y = pk_bf(x0.z, x0.w);
        px.z = pk_bf(x1.x, x1.y); px.w = pk_bf(x1.z, x1.w);
        ph.x = pk_bf(h0.x, h0.y); ph.y = pk_bf(h0.z, h0.w);
        ph.z = pk_bf(h1.x, h1.y); ph.w = pk_bf(h1.z, h1.w);
        int base = r * 128 + db;                 // ushort index; *2 bytes -> 16B aligned
        *(uint4*)&Ax[base] = px;
        *(uint4*)&Ah[base] = ph;
    }
    __syncthreads();

    const int wave = t >> 5;
    const int lane = t & 31;
    const int m    = lane & 15;
    const int koff = (lane < 16) ? 0 : 16;

    // ---- A fragments for all 4 K-steps (x and h), contiguous k = 16*(lane/16) + s, same as Wf ----
    v16bf ax[4], ah[4];
#pragma unroll
    for (int kt = 0; kt < 4; ++kt) {
        v8i tx, th;
#pragma unroll
        for (int v = 0; v < 8; ++v) {
            int k = kt * 32 + koff + 2 * v;
            tx[v] = *(const int*)&Ax[m * 128 + k];
            th[v] = *(const int*)&Ah[m * 128 + k];
        }
        ax[kt] = as_v16bf(tx);
        ah[kt] = as_v16bf(th);
    }

    // ---- 6 accumulators: (gx,gh) x (r,z,n); 24 WMMAs per wave ----
    v8f accx[3], acch[3];
#pragma unroll
    for (int gi = 0; gi < 3; ++gi) {
        int jt = gi * 8 + wave;             // col tile in the 384-wide gate matrix
        v8f cx = {}, ch = {};
#pragma unroll
        for (int kt = 0; kt < 4; ++kt) {
            v16bf bx = load_bfrag(WihF, jt, kt, lane);
            cx = __builtin_amdgcn_wmma_f32_16x16x32_bf16(
                false, ax[kt], false, bx, (short)0, cx, false, false);
            v16bf bh = load_bfrag(WhhF, jt, kt, lane);
            ch = __builtin_amdgcn_wmma_f32_16x16x32_bf16(
                false, ah[kt], false, bh, (short)0, ch, false, false);
        }
        accx[gi] = cx;
        acch[gi] = ch;
    }

    // ---- gates in registers (C/D layout: row = g + 8*(lane>=16), col = lane%16) ----
    const int d0  = wave * 16 + (lane & 15);
    const float bx0 = bih[0 * 128 + d0], bh0 = bhh[0 * 128 + d0];
    const float bx1 = bih[1 * 128 + d0], bh1 = bhh[1 * 128 + d0];
    const float bx2 = bih[2 * 128 + d0], bh2 = bhh[2 * 128 + d0];
    const int rb = row0 + ((lane < 16) ? 0 : 8);
#pragma unroll
    for (int g = 0; g < 8; ++g) {
        int row = rb + g;
        if (row < N) {
            long idx = (long)row * D_N + d0;
            float hv = h[idx];
            float xr = accx[0][g] + bx0, hr = acch[0][g] + bh0;
            float xz = accx[1][g] + bx1, hz = acch[1][g] + bh1;
            float xn = accx[2][g] + bx2, hn = acch[2][g] + bh2;
            float rg = 1.f / (1.f + __expf(-(xr + hr)));
            float zg = 1.f / (1.f + __expf(-(xz + hz)));
            float ng = tanhf(xn + rg * hn);
            h[idx] = (1.f - zg) * ng + zg * hv;
        }
    }
}

// ---------------------------------------------------------------- final FC
__global__ void __launch_bounds__(256)
fc_k(const float* __restrict__ es, const float* __restrict__ fw,
     const float* __restrict__ fb, float* __restrict__ out, int N) {
    __shared__ float W[C_N * D_N];
    __shared__ float B[C_N];
    int t = threadIdx.x;
    for (int i = t; i < C_N * D_N; i += 256) W[i] = fw[i];
    if (t < C_N) B[t] = fb[t];
    __syncthreads();
    int row = blockIdx.x * 8 + (t >> 5);
    int col = t & 31;
    if (row >= N) return;
    const float4* er = (const float4*)(es + (long)row * D_N);
    const float4* wr = (const float4*)(W + col * D_N);
    float s = B[col];
#pragma unroll
    for (int k = 0; k < D_N / 4; ++k) {
        float4 a = er[k], b = wr[k];
        s += a.x * b.x + a.y * b.y + a.z * b.z + a.w * b.w;
    }
    out[(long)row * C_N + col] = s;
}

// ================================================================ launch
extern "C" void kernel_launch(void* const* d_in, const int* in_sizes, int n_in,
                              void* d_out, int out_size, void* d_ws, size_t ws_size,
                              hipStream_t stream) {
    const int*   seed_index = (const int*)d_in[0];
    const float* es_in      = (const float*)d_in[1];
    const float* ps_in      = (const float*)d_in[2];
    const int*   erow       = (const int*)d_in[3];
    const int*   ecol       = (const int*)d_in[4];
    const float* evals      = (const float*)d_in[5];
    // weight matrices in order: l0eWih, l0eWhh, l0pWih, l0pWhh, l1eWih, l1eWhh, l1pWih, l1pWhh
    const float* Wsrc[8] = {
        (const float*)d_in[6],  (const float*)d_in[7],
        (const float*)d_in[10], (const float*)d_in[11],
        (const float*)d_in[14], (const float*)d_in[15],
        (const float*)d_in[18], (const float*)d_in[19]
    };
    const float* be_ih[2] = { (const float*)d_in[8],  (const float*)d_in[16] };
    const float* be_hh[2] = { (const float*)d_in[9],  (const float*)d_in[17] };
    const float* bp_ih[2] = { (const float*)d_in[12], (const float*)d_in[20] };
    const float* bp_hh[2] = { (const float*)d_in[13], (const float*)d_in[21] };
    const float* fc_w = (const float*)d_in[22];
    const float* fc_b = (const float*)d_in[23];

    // ---- workspace carve-up ----
    char* ws = (char*)d_ws;
    size_t off = 0;
    float* es_cur = (float*)(ws + off); off += (size_t)E_N * D_N * 4;
    float* ps_cur = (float*)(ws + off); off += (size_t)P_N * D_N * 4;
    float* e_msg  = (float*)(ws + off); off += (size_t)E_N * D_N * 4;
    float* p_msg  = (float*)(ws + off); off += (size_t)P_N * D_N * 4;
    float* mean   = (float*)(ws + off); off += 512;
    unsigned int* Wf[8];
    for (int i = 0; i < 8; ++i) { Wf[i] = (unsigned int*)(ws + off); off += (384 * 128 / 2) * 4; }

    const long e_f4 = (long)E_N * D_N / 4;   // 6.4M float4
    const long p_f4 = (long)P_N * D_N / 4;   // 1.6M float4

    // working copies of es / ps (inputs must stay pristine)
    copy_f4<<<25000, 256, 0, stream>>>((const float4*)es_in, (float4*)es_cur, e_f4);
    copy_f4<<<6250,  256, 0, stream>>>((const float4*)ps_in, (float4*)ps_cur, p_f4);

    // bf16 fragment-layout weights (tiny: 768 KB total, L2-resident)
    for (int i = 0; i < 8; ++i)
        conv_wfrag<<<96, 256, 0, stream>>>(Wsrc[i], Wf[i]);

    const int scat_blocks = (NNZ_N + 7) / 8;

    for (int l = 0; l < 2; ++l) {
        // seeds.mean from es at layer start
        seed_mean_k<<<1, 128, 0, stream>>>(seed_index, es_cur, mean);

        // p_msg = segment_sum(es[edge_rows] * w -> edge_cols)
        zero_f4<<<6250, 256, 0, stream>>>((float4*)p_msg, p_f4);
        scatter_add_k<<<scat_blocks, 256, 0, stream>>>(es_cur, erow, ecol, evals, p_msg, NNZ_N);

        // ps = GRU_p(p_msg, ps)
        gru_wmma_k<<<P_N / 16, 256, 0, stream>>>(p_msg, ps_cur,
                                                 Wf[l * 4 + 2], Wf[l * 4 + 3],
                                                 bp_ih[l], bp_hh[l], nullptr, P_N);

        // e_msg = segment_sum(ps[edge_cols] * w -> edge_rows)
        zero_f4<<<25000, 256, 0, stream>>>((float4*)e_msg, e_f4);
        scatter_add_k<<<scat_blocks, 256, 0, stream>>>(ps_cur, ecol, erow, evals, e_msg, NNZ_N);

        // es = GRU_e(e_msg + mean, es)
        gru_wmma_k<<<E_N / 16, 256, 0, stream>>>(e_msg, es_cur,
                                                 Wf[l * 4 + 0], Wf[l * 4 + 1],
                                                 be_ih[l], be_hh[l], mean, E_N);
    }

    // outputs: [logits (E*C)] [es (E*D)] [ps (P*D)]
    float* out   = (float*)d_out;
    float* o_es  = out + (long)E_N * C_N;
    float* o_ps  = o_es + (long)E_N * D_N;
    fc_k<<<E_N / 8, 256, 0, stream>>>(es_cur, fc_w, fc_b, out, E_N);
    copy_f4<<<25000, 256, 0, stream>>>((const float4*)es_cur, (float4*)o_es, e_f4);
    copy_f4<<<6250,  256, 0, stream>>>((const float4*)ps_cur, (float4*)o_ps, p_f4);
}
